// MultiHeadAttention_11141145166304
// MI455X (gfx1250) — matrix operand order repaired and checked by reading, not compiled, not run on hardware
//
#include <hip/hip_runtime.h>
#include <hip/hip_bf16.h>

// Problem constants (fixed by the reference)
#define Bsz 2
#define Sq  2048
#define Hh  16
#define Dd  128
#define DH  (Hh * Dd)      // 2048
#define Mrows (Bsz * Sq)   // 4096

typedef __bf16 bf16_t;
typedef __attribute__((ext_vector_type(16))) __bf16 bf16x16;
typedef __attribute__((ext_vector_type(8)))  __bf16 bf16x8;
typedef __attribute__((ext_vector_type(8)))  float  f32x8;

// ---- WMMA operand loaders (per CDNA5 ISA §7.12.2 VGPR layouts) ----
// A-matrix 16x32 bf16: lane L holds row M=L%16; elements [0..7]=K[base..base+7],
// [8..15]=K[base+16..base+23], base=(L/16)*8.
__device__ __forceinline__ bf16x16 load_a16(const bf16_t* row, int hi) {
  const bf16x8 lo = *(const bf16x8*)(row + hi * 8);
  const bf16x8 hh = *(const bf16x8*)(row + hi * 8 + 16);
  bf16x16 r;
#pragma unroll
  for (int i = 0; i < 8; ++i) { r[i] = lo[i]; r[i + 8] = hh[i]; }
  return r;
}
// B-matrix 32x16 bf16: lane L holds col N=L%16; elements [0..7]=K[base..base+7],
// [8..15]=K[base+16..base+23], base=(L/16)*8.  `row` points at B^T[N][k0].
__device__ __forceinline__ bf16x16 load_b16(const bf16_t* row, int hi) {
  const bf16x8 lo = *(const bf16x8*)(row + hi * 8);
  const bf16x8 hh = *(const bf16x8*)(row + hi * 8 + 16);
  bf16x16 r;
#pragma unroll
  for (int i = 0; i < 8; ++i) { r[i] = lo[i]; r[i + 8] = hh[i]; }
  return r;
}

__device__ __forceinline__ f32x8 wmma_bf16(bf16x16 a, bf16x16 b, f32x8 c) {
  return __builtin_amdgcn_wmma_f32_16x16x32_bf16(false, a, false, b,
                                                 (short)0, c, false, false);
}

// ---------------- conversion kernels ----------------
__global__ void cvt_copy_kernel(const float* __restrict__ src,
                                bf16_t* __restrict__ dst, int n) {
  int i = blockIdx.x * blockDim.x + threadIdx.x;
  if (i < n) dst[i] = (bf16_t)src[i];
}

// W is (K x N) row-major f32; WT is (N x K) row-major bf16.
__global__ void cvt_transpose_kernel(const float* __restrict__ W,
                                     bf16_t* __restrict__ WT, int K, int N) {
  int i = blockIdx.x * blockDim.x + threadIdx.x;
  if (i < K * N) {
    int k = i / N, n = i - k * N;
    WT[(size_t)n * K + k] = (bf16_t)W[i];
  }
}

// ---------------- QKV projection ----------------
// C = A(4096x128) @ W(128x2048) + bias, scaled; one wave per 16x16 tile.
// vmode==0: store [b][h][s][d] (Q with scale=1/sqrt(D), K with scale=1)
// vmode==1: store transposed  [b][h][d][s] (V), one b128 store per lane.
__global__ __launch_bounds__(256) void proj_kernel(
    const bf16_t* __restrict__ A, const bf16_t* __restrict__ BT,
    const float* __restrict__ bias, bf16_t* __restrict__ out,
    float scale, int vmode) {
  const int lane = threadIdx.x & 31, wid = threadIdx.x >> 5;
  const int tile = blockIdx.x * 8 + wid;        // 32768 tiles total
  const int tm = tile >> 7, tn = tile & 127;    // 256 x 128 tiles
  const int ln = lane & 15, hi = lane >> 4;

  const bf16_t* arow = A  + (size_t)(tm * 16 + ln) * Dd;
  const bf16_t* brow = BT + (size_t)(tn * 16 + ln) * Dd;
  f32x8 acc = {0.f,0.f,0.f,0.f,0.f,0.f,0.f,0.f};
#pragma unroll
  for (int c = 0; c < 4; ++c)
    acc = wmma_bf16(load_a16(arow + c * 32, hi), load_b16(brow + c * 32, hi), acc);

  const int n = tn * 16 + ln, h = n >> 7, d = n & 127;
  const float bv = bias[n];
  if (vmode) {                       // V^T: [b][h][d][s], s contiguous
    const int b  = (tm * 16) >> 11;
    const int s0 = tm * 16 + 8 * hi - b * Sq;
    bf16x8 pk;
#pragma unroll
    for (int r = 0; r < 8; ++r) pk[r] = (bf16_t)((acc[r] + bv) * scale);
    *(bf16x8*)(out + ((size_t)(b * Hh + h) * Dd + d) * Sq + s0) = pk;
  } else {                           // Q/K: [b][h][s][d]
#pragma unroll
    for (int r = 0; r < 8; ++r) {
      const int mm = tm * 16 + r + 8 * hi;
      const int b = mm >> 11, s = mm & (Sq - 1);
      out[((size_t)(b * Hh + h) * Sq + s) * Dd + d] = (bf16_t)((acc[r] + bv) * scale);
    }
  }
}

// ---------------- flash attention ----------------
// One wave per (b,h, 16-row Q block). Q pre-scaled by 1/sqrt(D).
__global__ __launch_bounds__(256) void attn_kernel(
    const bf16_t* __restrict__ Qg, const bf16_t* __restrict__ Kg,
    const bf16_t* __restrict__ Vt, bf16_t* __restrict__ AO) {
  __shared__ bf16_t plds[8 * 16 * 32];          // 1KB per wave P-repack scratch
  const int lane = threadIdx.x & 31;
  const int wid  = threadIdx.x >> 5;
  const int gw   = blockIdx.x * 8 + wid;        // 0..4095
  const int qb   = gw & (Sq / 16 - 1);          // Q block
  const int bh   = gw >> 7;                     // b*H + h
  const int ln = lane & 15, hi = lane >> 4;

  const bf16_t* Qbh = Qg + (size_t)bh * Sq * Dd;
  const bf16_t* Kbh = Kg + (size_t)bh * Sq * Dd;
  const bf16_t* Vbh = Vt + (size_t)bh * Dd * Sq;

  bf16x16 Qa[4];
  const bf16_t* qrow = Qbh + (size_t)(qb * 16 + ln) * Dd;
#pragma unroll
  for (int c = 0; c < 4; ++c) Qa[c] = load_a16(qrow + c * 32, hi);

  f32x8 O[8];
  const f32x8 zero8 = {0.f,0.f,0.f,0.f,0.f,0.f,0.f,0.f};
#pragma unroll
  for (int dt = 0; dt < 8; ++dt) O[dt] = zero8;
  float m[8], l[8];
#pragma unroll
  for (int r = 0; r < 8; ++r) { m[r] = -3.0e38f; l[r] = 0.f; }

  bf16_t* pl = plds + wid * 512;

  for (int t0 = 0; t0 < Sq; t0 += 32) {
    // scores: two 16x16 sub-tiles over 32 keys
    f32x8 s0 = zero8, s1 = zero8;
    const bf16_t* k0 = Kbh + (size_t)(t0 + ln) * Dd;
    const bf16_t* k1 = Kbh + (size_t)(t0 + 16 + ln) * Dd;
#pragma unroll
    for (int c = 0; c < 4; ++c) {
      s0 = wmma_bf16(Qa[c], load_b16(k0 + c * 32, hi), s0);
      s1 = wmma_bf16(Qa[c], load_b16(k1 + c * 32, hi), s1);
    }
    // online softmax; row = (r, lane/16), 16 cols spread over half-wave lanes
    float corr[8];
#pragma unroll
    for (int r = 0; r < 8; ++r) {
      float mx = fmaxf(s0[r], s1[r]);
      mx = fmaxf(mx, __shfl_xor(mx, 1));
      mx = fmaxf(mx, __shfl_xor(mx, 2));
      mx = fmaxf(mx, __shfl_xor(mx, 4));
      mx = fmaxf(mx, __shfl_xor(mx, 8));
      const float mn = fmaxf(m[r], mx);
      const float cf = __expf(m[r] - mn);
      m[r] = mn;
      const float p0 = __expf(s0[r] - mn);
      const float p1 = __expf(s1[r] - mn);
      s0[r] = p0; s1[r] = p1;
      float rs = p0 + p1;
      rs += __shfl_xor(rs, 1);
      rs += __shfl_xor(rs, 2);
      rs += __shfl_xor(rs, 4);
      rs += __shfl_xor(rs, 8);
      l[r] = l[r] * cf + rs;
      corr[r] = cf;
    }
#pragma unroll
    for (int dt = 0; dt < 8; ++dt)
#pragma unroll
      for (int r = 0; r < 8; ++r) O[dt][r] *= corr[r];
    // repack P: C layout -> LDS -> A-operand layout (intra-wave, DS in-order)
#pragma unroll
    for (int r = 0; r < 8; ++r) {
      const int row = r + 8 * hi;
      pl[row * 32 + ln]      = (bf16_t)s0[r];
      pl[row * 32 + 16 + ln] = (bf16_t)s1[r];
    }
    bf16x16 Pa = load_a16(pl + ln * 32, hi);
    // O += P @ V : B operand = V^T rows (d), contiguous along t
#pragma unroll
    for (int dt = 0; dt < 8; ++dt) {
      bf16x16 vb = load_b16(Vbh + (size_t)(dt * 16 + ln) * Sq + t0, hi);
      O[dt] = wmma_bf16(Pa, vb, O[dt]);
    }
  }
  // normalize and store to AO[b][s][h*128+d] (bf16, row-major 4096x2048)
  const int b = bh >> 4, h = bh & 15;
#pragma unroll
  for (int r = 0; r < 8; ++r) l[r] = 1.0f / l[r];
#pragma unroll
  for (int dt = 0; dt < 8; ++dt)
#pragma unroll
    for (int r = 0; r < 8; ++r) {
      const int s = qb * 16 + r + 8 * hi;
      AO[((size_t)b * Sq + s) * DH + h * Dd + dt * 16 + ln] =
          (bf16_t)(O[dt][r] * l[r]);
    }
}

// ---------------- output projection ----------------
// out(4096x128) = AO(4096x2048) @ Wo(2048x128) + bo, f32 store.
__global__ __launch_bounds__(256) void outproj_kernel(
    const bf16_t* __restrict__ AO, const bf16_t* __restrict__ WoT,
    const float* __restrict__ bo, float* __restrict__ out) {
  const int lane = threadIdx.x & 31, wid = threadIdx.x >> 5;
  const int tile = blockIdx.x * 8 + wid;   // 2048 tiles
  const int tm = tile >> 3, tn = tile & 7; // 256 x 8
  const int ln = lane & 15, hi = lane >> 4;
  const bf16_t* arow = AO  + (size_t)(tm * 16 + ln) * DH;
  const bf16_t* brow = WoT + (size_t)(tn * 16 + ln) * DH;
  f32x8 acc = {0.f,0.f,0.f,0.f,0.f,0.f,0.f,0.f};
#pragma unroll
  for (int c = 0; c < 64; ++c)
    acc = wmma_bf16(load_a16(arow + c * 32, hi), load_b16(brow + c * 32, hi), acc);
  const float bv = bo[tn * 16 + ln];
#pragma unroll
  for (int r = 0; r < 8; ++r)
    out[(size_t)(tm * 16 + r + 8 * hi) * Dd + tn * 16 + ln] = acc[r] + bv;
}

// ---------------- host launch ----------------
extern "C" void kernel_launch(void* const* d_in, const int* in_sizes, int n_in,
                              void* d_out, int out_size, void* d_ws, size_t ws_size,
                              hipStream_t stream) {
  (void)in_sizes; (void)n_in; (void)out_size; (void)ws_size;
  const float* x  = (const float*)d_in[0];
  const float* Wq = (const float*)d_in[1];
  const float* bq = (const float*)d_in[2];
  const float* Wk = (const float*)d_in[3];
  const float* bk = (const float*)d_in[4];
  const float* Wv = (const float*)d_in[5];
  const float* bv = (const float*)d_in[6];
  const float* Wo = (const float*)d_in[7];
  const float* bo = (const float*)d_in[8];
  float* out = (float*)d_out;

  char* ws = (char*)d_ws;
  size_t off = 0;
  auto alloc = [&](size_t bytes) { void* p = ws + off; off += bytes; return p; };
  bf16_t* xb  = (bf16_t*)alloc((size_t)Mrows * Dd * 2);   // 1 MB
  bf16_t* WqT = (bf16_t*)alloc((size_t)DH * Dd * 2);      // 0.5 MB
  bf16_t* WkT = (bf16_t*)alloc((size_t)DH * Dd * 2);
  bf16_t* WvT = (bf16_t*)alloc((size_t)DH * Dd * 2);
  bf16_t* WoT = (bf16_t*)alloc((size_t)Dd * DH * 2);
  bf16_t* Qg  = (bf16_t*)alloc((size_t)Bsz * Hh * Sq * Dd * 2);  // 16 MB
  bf16_t* Kg  = (bf16_t*)alloc((size_t)Bsz * Hh * Sq * Dd * 2);
  bf16_t* Vtg = (bf16_t*)alloc((size_t)Bsz * Hh * Dd * Sq * 2);
  bf16_t* AOg = (bf16_t*)alloc((size_t)Mrows * DH * 2);          // 16 MB

  cvt_copy_kernel<<<(Mrows * Dd + 255) / 256, 256, 0, stream>>>(x, xb, Mrows * Dd);
  cvt_transpose_kernel<<<(Dd * DH + 255) / 256, 256, 0, stream>>>(Wq, WqT, Dd, DH);
  cvt_transpose_kernel<<<(Dd * DH + 255) / 256, 256, 0, stream>>>(Wk, WkT, Dd, DH);
  cvt_transpose_kernel<<<(Dd * DH + 255) / 256, 256, 0, stream>>>(Wv, WvT, Dd, DH);
  cvt_transpose_kernel<<<(DH * Dd + 255) / 256, 256, 0, stream>>>(Wo, WoT, DH, Dd);

  const float qscale = 0.08838834764831845f;  // 1/sqrt(128), baked into Q
  proj_kernel<<<4096, 256, 0, stream>>>(xb, WqT, bq, Qg, qscale, 0);
  proj_kernel<<<4096, 256, 0, stream>>>(xb, WkT, bk, Kg, 1.0f, 0);
  proj_kernel<<<4096, 256, 0, stream>>>(xb, WvT, bv, Vtg, 1.0f, 1);
  attn_kernel<<<512, 256, 0, stream>>>(Qg, Kg, Vtg, AOg);
  outproj_kernel<<<256, 256, 0, stream>>>(AOg, WoT, bo, out);
}
